// HeteroGCNEncoder_7791070675396
// MI455X (gfx1250) — matrix operand order repaired and checked by reading, not compiled, not run on hardware
//
#include <hip/hip_runtime.h>
#include <hip/hip_bf16.h>

typedef __attribute__((ext_vector_type(16))) _Float16 v16h;
typedef __attribute__((ext_vector_type(8)))  float    v8f;
typedef _Float16 half_t;

#define DN 128          // feature width D
#define EPS 1e-5f

static __host__ __device__ inline int cdiv(int a, int b) { return (a + b - 1) / b; }

// ---------------------------------------------------------------------------
// Column statistics: sum and sum-of-squares per column (for BatchNorm fold).
// mode 0: X[r,c] = in0[r*din + c]                       (din == 128)
// mode 1: X[r,c] = in0[src[r]*128 + c] + in1[dst[r]*128 + c]
// mode 2: X[r,c] = (c < 128) ? in0[r*128+c] : in1[r*128+(c-128)]   (din == 256)
// blockDim = 128; one row per loop iteration (coalesced 512B reads).
// ---------------------------------------------------------------------------
__global__ __launch_bounds__(128)
void colstats_kernel(const float* __restrict__ in0, const float* __restrict__ in1,
                     const int* __restrict__ src, const int* __restrict__ dst,
                     int rows, int din, int mode,
                     float* __restrict__ colsum, float* __restrict__ colsumsq) {
  int t = threadIdx.x;                 // column 0..127
  int two = (din == 2 * DN) ? 1 : 0;
  float s0 = 0.f, q0 = 0.f, s1 = 0.f, q1 = 0.f;
  for (int r = blockIdx.x; r < rows; r += gridDim.x) {
    if (mode == 1) {
      int a = src[r], b = dst[r];
      float x = in0[(long)a * DN + t] + in1[(long)b * DN + t];
      s0 += x; q0 += x * x;
    } else if (mode == 2) {
      float x0 = in0[(long)r * DN + t];
      float x1 = in1[(long)r * DN + t];
      s0 += x0; q0 += x0 * x0;
      s1 += x1; q1 += x1 * x1;
    } else {
      float x = in0[(long)r * din + t];
      s0 += x; q0 += x * x;
    }
  }
  atomicAdd(&colsum[t], s0);
  atomicAdd(&colsumsq[t], q0);
  if (two) {
    atomicAdd(&colsum[t + DN], s1);
    atomicAdd(&colsumsq[t + DN], q1);
  }
}

// ---------------------------------------------------------------------------
// Fold BN into the linear layer:
//   s_i = g_i * rsqrt(var_i + eps);  t_i = b_i - mean_i * s_i
//   W'[i,j] = s_i * W[i,j]  (stored f16 for WMMA)
//   bias'[j] = bias[j] + sum_i t_i * W[i,j]   (accumulated via atomics)
// grid.x = din, blockDim = 128 (j)
// ---------------------------------------------------------------------------
__global__ __launch_bounds__(128)
void fuse_kernel(const float* __restrict__ colsum, const float* __restrict__ colsumsq,
                 float rowsInv,
                 const float* __restrict__ g, const float* __restrict__ b,
                 const float* __restrict__ W, const float* __restrict__ bias,
                 int din, half_t* __restrict__ Wf, float* __restrict__ biasf) {
  int i = blockIdx.x;
  int j = threadIdx.x;
  float m  = colsum[i] * rowsInv;
  float v  = colsumsq[i] * rowsInv - m * m;
  float sc = g[i] * rsqrtf(v + EPS);
  float t  = b[i] - m * sc;
  float w  = W[(long)i * DN + j];
  Wf[(long)i * DN + j] = (half_t)(sc * w);
  atomicAdd(&biasf[j], t * w);
  if (i == 0) atomicAdd(&biasf[j], bias[j]);
}

// ---------------------------------------------------------------------------
// In-degree count per node (float so the divide kernel is trivial).
// ---------------------------------------------------------------------------
__global__ void deg_kernel(const int* __restrict__ dst, int E, float* __restrict__ deg) {
  int e = blockIdx.x * blockDim.x + threadIdx.x;
  if (e < E) atomicAdd(&deg[dst[e]], 1.0f);
}

// ---------------------------------------------------------------------------
// h = 0.5 * (acc_win / max(deg_win,1) + acc_loss / max(deg_loss,1)); optional f=h
// ---------------------------------------------------------------------------
__global__ void combine_kernel(const float* __restrict__ accw, const float* __restrict__ accl,
                               const float* __restrict__ degw, const float* __restrict__ degl,
                               int N, float* __restrict__ h, float* __restrict__ f) {
  int i = blockIdx.x * blockDim.x + threadIdx.x;
  if (i >= N * DN) return;
  int n = i >> 7;
  float dw = fmaxf(degw[n], 1.0f);
  float dl = fmaxf(degl[n], 1.0f);
  float v  = 0.5f * (accw[i] / dw + accl[i] / dl);
  h[i] = v;
  if (f) f[i] = v;
}

// ---------------------------------------------------------------------------
// A-fragment loader: 16-bit A 16x32, contiguous K order (same as B).
//   low lanes (hi=0): K = k0+0..7  (elems 0..7),  k0+8..15 (elems 8..15)
//   high lanes (hi=1): K = k0+16..23,             k0+24..31
// ---------------------------------------------------------------------------
template <int MODE_IN>
__device__ __forceinline__ v16h load_a_frag(const float* __restrict__ pa0,
                                            const float* __restrict__ pa1,
                                            int k0, int hi) {
  v16h a;
  int kA = k0 + hi * 16;
#pragma unroll
  for (int hh = 0; hh < 2; ++hh) {
    int kk = kA + hh * 8;
    const float* p;
    int off;
    if (MODE_IN == 2) { p = (kk < DN) ? pa0 : pa1; off = kk & (DN - 1); }
    else              { p = pa0;                   off = kk; }
    float xv[8];
    *(float4*)&xv[0] = *(const float4*)(p + off);
    *(float4*)&xv[4] = *(const float4*)(p + off + 4);
    if (MODE_IN == 1) {
      float yv[8];
      *(float4*)&yv[0] = *(const float4*)(pa1 + off);
      *(float4*)&yv[4] = *(const float4*)(pa1 + off + 4);
#pragma unroll
      for (int q = 0; q < 8; ++q) xv[q] += yv[q];
    }
#pragma unroll
    for (int q = 0; q < 8; ++q) a[hh * 8 + q] = (half_t)xv[q];
  }
  return a;
}

// ---------------------------------------------------------------------------
// Fused WMMA GEMM:  Y = relu( X @ W'(f16) + bias' )
//   MODE_IN  0: X row r = in0 + r*din                (din = 128)
//   MODE_IN  1: X row r = in0[gsrc[r]] + in1[gdst[r]] (gather-add, din = 128)
//   MODE_IN  2: X row r = concat(in0[r], in1[r])     (din = 256)
//   MODE_OUT 0: out[r*128 + c] = y
//   MODE_OUT 1: atomicAdd(&out[sdst[r]*128 + c], y)  (fused scatter for seg-mean)
//
// Block = 128 threads = 4 waves; each wave computes TWO 16-row strips
// (32 rows) x 128 cols, so every B fragment fetched from LDS feeds two
// back-to-back v_wmma_f32_16x16x32_f16 ops (halves LDS traffic per FLOP and
// covers the WMMA->WMMA hazard window with independent work).
// Weights live in LDS transposed [n][k] so each lane's B fragment is one
// contiguous 32B vector (16 f16) load.
// ---------------------------------------------------------------------------
template <int MODE_IN, int MODE_OUT>
__global__ __launch_bounds__(128)
void gemm_kernel(const float* __restrict__ in0, const float* __restrict__ in1,
                 const int* __restrict__ gsrc, const int* __restrict__ gdst,
                 const half_t* __restrict__ Wf, const float* __restrict__ biasf,
                 int rows, int din,
                 float* __restrict__ out, const int* __restrict__ sdst) {
  __shared__ __align__(32) half_t sWT[2 * DN * DN]; // [n][k] transposed, 64KB max

  int tid = threadIdx.x;
  // Stage W' into LDS, transposing [k][n] -> [n][k].
  for (int idx = tid; idx < din * DN; idx += 128) {
    int k = idx / DN, n = idx % DN;
    sWT[n * din + k] = Wf[idx];
  }
  __syncthreads();

  int lane  = tid & 31;
  int wave  = tid >> 5;
  int mlane = lane & 15;   // A row; B/C column within a 16-wide tile
  int hi    = lane >> 4;   // half-wave selector

  long rowbase = (long)blockIdx.x * 128 + wave * 32;   // 2 strips per wave

  // Per-strip A row pointers (clamped for the ragged tail).
  const float* p0[2];
  const float* p1[2] = {nullptr, nullptr};
#pragma unroll
  for (int s = 0; s < 2; ++s) {
    long arow   = rowbase + s * 16 + mlane;
    long rclamp = (arow < rows) ? arow : (long)(rows - 1);
    if (MODE_IN == 1) {
      p0[s] = in0 + (long)gsrc[rclamp] * DN;
      p1[s] = in1 + (long)gdst[rclamp] * DN;
    } else if (MODE_IN == 2) {
      p0[s] = in0 + rclamp * DN;
      p1[s] = in1 + rclamp * DN;
    } else {
      p0[s] = in0 + rclamp * din;
    }
  }

  v8f acc[2][8];
#pragma unroll
  for (int s = 0; s < 2; ++s)
#pragma unroll
    for (int n = 0; n < 8; ++n)
#pragma unroll
      for (int q = 0; q < 8; ++q) acc[s][n][q] = 0.0f;

  for (int k0 = 0; k0 < din; k0 += 32) {
    v16h a0 = load_a_frag<MODE_IN>(p0[0], p1[0], k0, hi);
    v16h a1 = load_a_frag<MODE_IN>(p0[1], p1[1], k0, hi);

    int kb = k0 + hi * 16;  // low lanes K 0..15, high lanes K 16..31 of this step
#pragma unroll
    for (int n = 0; n < 8; ++n) {
      int col = n * 16 + mlane;
      v16h bfrag = *(const v16h*)&sWT[col * din + kb];
      acc[0][n] = __builtin_amdgcn_wmma_f32_16x16x32_f16(
          false, a0, false, bfrag, (short)0, acc[0][n], false, false);
      acc[1][n] = __builtin_amdgcn_wmma_f32_16x16x32_f16(
          false, a1, false, bfrag, (short)0, acc[1][n], false, false);
    }
  }

  // ---- Epilogue: bias + ReLU, then store or atomic scatter-add.
#pragma unroll
  for (int s = 0; s < 2; ++s) {
    long sbase = rowbase + s * 16;
    int dstc[8];
    if (MODE_OUT == 1) {
#pragma unroll
      for (int v = 0; v < 8; ++v) {
        long r = sbase + v + 8 * hi;
        dstc[v] = (r < rows) ? sdst[r] : 0;
      }
    }
#pragma unroll
    for (int n = 0; n < 8; ++n) {
      int col  = n * 16 + mlane;
      float bv = biasf[col];
#pragma unroll
      for (int v = 0; v < 8; ++v) {
        long r = sbase + v + 8 * hi;   // C layout: VGPR v -> M = v + 8*hi
        if (r >= rows) continue;
        float val = fmaxf(acc[s][n][v] + bv, 0.0f);
        if (MODE_OUT == 1) atomicAdd(&out[(long)dstc[v] * DN + col], val);
        else               out[r * DN + col] = val;
      }
    }
  }
}

// ---------------------------------------------------------------------------
// Host orchestration
// ---------------------------------------------------------------------------
extern "C" void kernel_launch(void* const* d_in, const int* in_sizes, int n_in,
                              void* d_out, int out_size, void* d_ws, size_t ws_size,
                              hipStream_t stream) {
  // Input order (see setup_inputs): win_f, loss_f, then 7 blocks x (g,b,W,bias),
  // then win_src, win_dst, loss_src, loss_dst, n_nodes.
  const float* win_f  = (const float*)d_in[0];
  const float* loss_f = (const float*)d_in[1];
  const float* P[7][4]; // wi, li, src, dst, wl, ll, out : g,b,W,bias
  for (int blk = 0; blk < 7; ++blk)
    for (int k = 0; k < 4; ++k) P[blk][k] = (const float*)d_in[2 + blk * 4 + k];
  const int* win_src  = (const int*)d_in[30];
  const int* win_dst  = (const int*)d_in[31];
  const int* loss_src = (const int*)d_in[32];
  const int* loss_dst = (const int*)d_in[33];

  const int Ew = in_sizes[30];
  const int El = in_sizes[32];
  const int N  = out_size / DN;

  // Workspace carve-up (floats).
  float* ws = (float*)d_ws;
  size_t o = 0;
  float* f_buf  = ws + o; o += (size_t)N * DN;
  float* h_buf  = ws + o; o += (size_t)N * DN;
  float* hu     = ws + o; o += (size_t)N * DN;
  float* hv     = ws + o; o += (size_t)N * DN;
  float* accw   = ws + o; o += (size_t)N * DN;
  float* accl   = ws + o; o += (size_t)N * DN;
  float* degw   = ws + o; o += (size_t)N;
  float* degl   = ws + o; o += (size_t)N;
  float* colsum = ws + o; o += 256;          // colsum(256) + colsumsq(256) contiguous
  float* colsq  = ws + o; o += 256;
  float* biasf  = ws + o; o += DN;
  half_t* Wf    = (half_t*)(ws + o);          // 2*128*128 f16 = 64KB

  const int SB = 512;  // colstats grid

  auto stats = [&](const float* i0, const float* i1, const int* s, const int* dd,
                   int rows, int din, int mode) {
    hipMemsetAsync(colsum, 0, 512 * sizeof(float), stream);
    colstats_kernel<<<SB, 128, 0, stream>>>(i0, i1, s, dd, rows, din, mode, colsum, colsq);
  };
  auto fuse = [&](int blk, int rows, int din) {
    hipMemsetAsync(biasf, 0, DN * sizeof(float), stream);
    fuse_kernel<<<din, 128, 0, stream>>>(colsum, colsq, 1.0f / (float)rows,
                                         P[blk][0], P[blk][1], P[blk][2], P[blk][3],
                                         din, Wf, biasf);
  };

  // --- degrees (once per launch) ---
  hipMemsetAsync(degw, 0, (size_t)2 * N * sizeof(float), stream);
  deg_kernel<<<cdiv(Ew, 256), 256, 0, stream>>>(win_dst, Ew, degw);
  deg_kernel<<<cdiv(El, 256), 256, 0, stream>>>(loss_dst, El, degl);

  // --- edge input blocks: BN+Linear+ReLU fused, scatter straight into acc ---
  stats(win_f, nullptr, nullptr, nullptr, Ew, DN, 0);
  fuse(0, Ew, DN); // wi
  hipMemsetAsync(accw, 0, (size_t)N * DN * sizeof(float), stream);
  gemm_kernel<0, 1><<<cdiv(Ew, 128), 128, 0, stream>>>(
      win_f, nullptr, nullptr, nullptr, Wf, biasf, Ew, DN, accw, win_dst);

  stats(loss_f, nullptr, nullptr, nullptr, El, DN, 0);
  fuse(1, El, DN); // li
  hipMemsetAsync(accl, 0, (size_t)N * DN * sizeof(float), stream);
  gemm_kernel<0, 1><<<cdiv(El, 128), 128, 0, stream>>>(
      loss_f, nullptr, nullptr, nullptr, Wf, biasf, El, DN, accl, loss_dst);

  combine_kernel<<<cdiv(N * DN, 256), 256, 0, stream>>>(accw, accl, degw, degl, N,
                                                        h_buf, f_buf);

  // --- message-passing depth = 3 ---
  for (int it = 0; it < 3; ++it) {
    stats(h_buf, nullptr, nullptr, nullptr, N, DN, 0);  // shared by src & dst blocks
    fuse(2, N, DN); // src
    gemm_kernel<0, 0><<<cdiv(N, 128), 128, 0, stream>>>(
        h_buf, nullptr, nullptr, nullptr, Wf, biasf, N, DN, hu, nullptr);
    fuse(3, N, DN); // dst
    gemm_kernel<0, 0><<<cdiv(N, 128), 128, 0, stream>>>(
        h_buf, nullptr, nullptr, nullptr, Wf, biasf, N, DN, hv, nullptr);

    // win edges: gather(hu,hv) -> BN+Linear+ReLU -> scatter-add
    stats(hu, hv, win_src, win_dst, Ew, DN, 1);
    fuse(4, Ew, DN); // wl
    hipMemsetAsync(accw, 0, (size_t)N * DN * sizeof(float), stream);
    gemm_kernel<1, 1><<<cdiv(Ew, 128), 128, 0, stream>>>(
        hu, hv, win_src, win_dst, Wf, biasf, Ew, DN, accw, win_dst);

    // loss edges
    stats(hu, hv, loss_src, loss_dst, El, DN, 1);
    fuse(5, El, DN); // ll
    hipMemsetAsync(accl, 0, (size_t)N * DN * sizeof(float), stream);
    gemm_kernel<1, 1><<<cdiv(El, 128), 128, 0, stream>>>(
        hu, hv, loss_src, loss_dst, Wf, biasf, El, DN, accl, loss_dst);

    combine_kernel<<<cdiv(N * DN, 256), 256, 0, stream>>>(accw, accl, degw, degl, N,
                                                          h_buf, nullptr);
  }

  // --- output block on concat([f, h]) with din = 256 ---
  stats(f_buf, h_buf, nullptr, nullptr, N, 2 * DN, 2);
  fuse(6, N, 2 * DN); // out
  gemm_kernel<2, 0><<<cdiv(N, 128), 128, 0, stream>>>(
      f_buf, h_buf, nullptr, nullptr, Wf, biasf, N, 2 * DN, (float*)d_out, nullptr);
}
